// LucidrainsSparseAttention_80204219285612
// MI455X (gfx1250) — matrix operand order repaired and checked by reading, not compiled, not run on hardware
//
#include <hip/hip_runtime.h>

#define H_   8
#define DH_  64
#define BC_  32
#define S_   4
#define W_   256
#define N_   1024
#define DIM_ 512
#define NB_  32      // N_/BC_
#define HD_  512     // H_*DH_

typedef __attribute__((ext_vector_type(16))) __bf16          bf16x16;
typedef __attribute__((ext_vector_type(16))) unsigned short  u16x16;
typedef __attribute__((ext_vector_type(8)))  unsigned short  u16x8;
typedef __attribute__((ext_vector_type(8)))  float           f32x8;

__device__ __forceinline__ unsigned short f2bf(float f) {
    unsigned int u = __builtin_bit_cast(unsigned int, f);
    u += 0x7fffu + ((u >> 16) & 1u);          // round-to-nearest-even
    return (unsigned short)(u >> 16);
}
__device__ __forceinline__ float bf2f(unsigned short s) {
    unsigned int u = ((unsigned int)s) << 16;
    return __builtin_bit_cast(float, u);
}

__device__ __forceinline__ f32x8 wmma_bf16(bf16x16 a, bf16x16 b, f32x8 c) {
    return __builtin_amdgcn_wmma_f32_16x16x32_bf16(false, a, false, b, (short)0, c, false, false);
}

// A fragment (16x32 bf16), row-major source. Lane: m = lane&15, hi = lane>>4.
// elements 0..7 = k(hi*8 + 0..7), elements 8..15 = k(16 + hi*8 + 0..7).
__device__ __forceinline__ bf16x16 load_a_frag(const unsigned short* rowm, int hi) {
    u16x8 lo  = *(const u16x8*)(rowm + hi * 8);
    u16x8 hi8 = *(const u16x8*)(rowm + 16 + hi * 8);
    u16x16 t;
#pragma unroll
    for (int e = 0; e < 8; ++e) { t[e] = lo[e]; t[e + 8] = hi8[e]; }
    return __builtin_bit_cast(bf16x16, t);
}

// B fragment (32x16 bf16) given B^T row-major [n][k]. Lane: n = lane&15, hi = lane>>4.
// elements 0..7 = k(hi*8 + 0..7), elements 8..15 = k(16 + hi*8 + 0..7).
__device__ __forceinline__ bf16x16 load_b_frag(const unsigned short* rown, int hi) {
    u16x8 lo  = *(const u16x8*)(rown + hi * 8);
    u16x8 hi8 = *(const u16x8*)(rown + 16 + hi * 8);
    u16x16 t;
#pragma unroll
    for (int e = 0; e < 8; ++e) { t[e] = lo[e]; t[e + 8] = hi8[e]; }
    return __builtin_bit_cast(bf16x16, t);
}

// ---------------- prep: x + pos@Wp + bp, RMSNorm ----------------
__global__ void prep_kernel(const float* __restrict__ x, const float* __restrict__ pos,
                            const float* __restrict__ gamma, const float* __restrict__ Wp,
                            const float* __restrict__ bp,
                            float* __restrict__ hf, unsigned short* __restrict__ hbf) {
    int i = blockIdx.x;
    int t = threadIdx.x;
    __shared__ float red[256];
    float p0 = pos[i * 3 + 0], p1 = pos[i * 3 + 1], p2 = pos[i * 3 + 2];
    int c0 = t * 2, c1 = t * 2 + 1;
    float v0 = x[(size_t)i * DIM_ + c0] + p0 * Wp[c0] + p1 * Wp[DIM_ + c0] + p2 * Wp[2 * DIM_ + c0] + bp[c0];
    float v1 = x[(size_t)i * DIM_ + c1] + p0 * Wp[c1] + p1 * Wp[DIM_ + c1] + p2 * Wp[2 * DIM_ + c1] + bp[c1];
    red[t] = v0 * v0 + v1 * v1;
    __syncthreads();
    for (int s = 128; s > 0; s >>= 1) {
        if (t < s) red[t] += red[t + s];
        __syncthreads();
    }
    float r = rsqrtf(red[0] * (1.0f / DIM_) + 1e-6f);
    float o0 = v0 * r * gamma[c0], o1 = v1 * r * gamma[c1];
    hf[(size_t)i * DIM_ + c0] = o0;       hf[(size_t)i * DIM_ + c1] = o1;
    hbf[(size_t)i * DIM_ + c0] = f2bf(o0); hbf[(size_t)i * DIM_ + c1] = f2bf(o1);
}

// ---------------- transpose f32 (K x N) -> bf16 (N x K) ----------------
__global__ void transpose_bf16_kernel(const float* __restrict__ W, unsigned short* __restrict__ WT,
                                      int K, int Ncol) {
    int id = blockIdx.x * blockDim.x + threadIdx.x;
    if (id >= K * Ncol) return;
    int k = id / Ncol, n = id % Ncol;
    WT[(size_t)n * K + k] = f2bf(W[id]);
}

// ---------------- generic WMMA GEMM: C(MxN,f32) = A(MxK,bf16) * BT(NxK,bf16)^T (+bias) ----
__global__ void gemm_bf16_wmma_kernel(const unsigned short* __restrict__ A,
                                      const unsigned short* __restrict__ BT,
                                      const float* __restrict__ bias,
                                      float* __restrict__ C, int M, int Ncol, int K) {
    int wid  = (blockIdx.x * blockDim.x + threadIdx.x) >> 5;
    int lane = threadIdx.x & 31;
    int tilesN = Ncol >> 4;
    int tiles  = (M >> 4) * tilesN;
    if (wid >= tiles) return;
    int tm = wid / tilesN, tn = wid % tilesN;
    int lm = lane & 15, hi = lane >> 4;
    const unsigned short* arow = A  + (size_t)(tm * 16 + lm) * K;
    const unsigned short* brow = BT + (size_t)(tn * 16 + lm) * K;
    f32x8 acc = {};
    for (int k0 = 0; k0 < K; k0 += 32) {
        __builtin_prefetch(arow + k0 + 256, 0, 1);
        bf16x16 af = load_a_frag(arow + k0, hi);
        bf16x16 bfv = load_b_frag(brow + k0, hi);
        acc = wmma_bf16(af, bfv, acc);
    }
    int col = tn * 16 + lm;
    float bv = bias ? bias[col] : 0.0f;
#pragma unroll
    for (int r = 0; r < 8; ++r) {
        int row = tm * 16 + r + 8 * hi;
        C[(size_t)row * Ncol + col] = acc[r] + bv;
    }
}

// ---------------- repack q/k/v into per-head bf16 layouts ----------------
__global__ void qkv_repack_kernel(const float* __restrict__ qf, const float* __restrict__ kf,
                                  const float* __restrict__ vf,
                                  const float* __restrict__ k_pos, const float* __restrict__ v_pos,
                                  unsigned short* __restrict__ qh, unsigned short* __restrict__ kh,
                                  unsigned short* __restrict__ vh, unsigned short* __restrict__ vT,
                                  unsigned short* __restrict__ kbflat, unsigned short* __restrict__ vbflat) {
    int id = blockIdx.x * blockDim.x + threadIdx.x;
    if (id >= N_ * HD_) return;
    int i = id >> 9, c = id & (HD_ - 1);
    int hh = c >> 6, d = c & 63;
    float qv = qf[id], kv = kf[id], vv = vf[id];
    size_t hoff = ((size_t)hh * N_ + i) * DH_ + d;
    qh[hoff] = f2bf(qv);
    kh[hoff] = f2bf(kv);
    vh[hoff] = f2bf(vv);
    vT[(size_t)hh * DH_ * N_ + (size_t)d * N_ + i] = f2bf(vv);
    int blk = i >> 5, cp = i & 31;
    size_t bo = ((size_t)hh * NB_ + blk) * (BC_ * DH_) + (size_t)cp * DH_ + d;
    int poff = (hh * BC_ + cp) * DH_ + d;
    kbflat[bo] = f2bf(kv + k_pos[poff]);
    vbflat[bo] = f2bf(vv + v_pos[poff]);
}

__global__ void cvt_bf16_kernel(const float* __restrict__ src, unsigned short* __restrict__ dst, int count) {
    int id = blockIdx.x * blockDim.x + threadIdx.x;
    if (id < count) dst[id] = f2bf(src[id]);
}

// ---------------- compressed branch: WMMA scores + per-lane softmax/out_c/top-4 ----------------
__global__ void compressed_kernel(const unsigned short* __restrict__ qh, const unsigned short* __restrict__ ckbf,
                                  const float* __restrict__ cvf, const float* __restrict__ mem_ck,
                                  const float* __restrict__ mem_cv,
                                  float* __restrict__ out_c, int* __restrict__ sel) {
    __shared__ float sS[4][16 * NB_];
    int wid  = (blockIdx.x * blockDim.x + threadIdx.x) >> 5;
    int lane = threadIdx.x & 31;
    int wib  = (threadIdx.x >> 5) & 3;
    if (wid >= H_ * (N_ / 16)) return;
    int h = wid >> 6, qt = wid & 63, qlo = qt * 16;
    int lm = lane & 15, hi = lane >> 4;
    const unsigned short* qrow = qh + ((size_t)h * N_ + qlo + lm) * DH_;
    bf16x16 a0 = load_a_frag(qrow, hi);
    bf16x16 a1 = load_a_frag(qrow + 32, hi);
#pragma unroll
    for (int ct = 0; ct < 2; ++ct) {
        const unsigned short* brow = ckbf + ((size_t)h * NB_ + ct * 16 + lm) * DH_;
        f32x8 s = {};
        s = wmma_bf16(a0, load_b_frag(brow, hi), s);
        s = wmma_bf16(a1, load_b_frag(brow + 32, hi), s);
#pragma unroll
        for (int r = 0; r < 8; ++r) sS[wib][(r + 8 * hi) * NB_ + ct * 16 + lm] = s[r];
    }
    // same-wave DS ops are in-order (ISA 7.3) -> safe to read back without barrier
    if (lane < 16) {
        int i = qlo + lane;
        const unsigned short* qp = qh + ((size_t)h * N_ + i) * DH_;
        float qreg[DH_];
#pragma unroll
        for (int d = 0; d < DH_; ++d) qreg[d] = bf2f(qp[d]);
        float ms = 0.0f;
#pragma unroll
        for (int d = 0; d < DH_; ++d) ms += qreg[d] * mem_ck[h * DH_ + d];
        ms *= 0.125f;
        float e[NB_];
        float mx = ms;
        for (int j = 0; j < NB_; ++j) {
            float v = (i >= (j + 1) * BC_ - 1) ? sS[wib][lane * NB_ + j] * 0.125f : -1e30f;
            e[j] = v;
            mx = fmaxf(mx, v);
        }
        float em = __expf(ms - mx);
        float esum = em;
        for (int j = 0; j < NB_; ++j) {
            float t = (e[j] > -1e29f) ? __expf(e[j] - mx) : 0.0f;
            e[j] = t;
            esum += t;
        }
        float inv = 1.0f / esum;
        float acc[DH_];
        float wm = em * inv;
#pragma unroll
        for (int d = 0; d < DH_; ++d) acc[d] = wm * mem_cv[h * DH_ + d];
        for (int j = 0; j < NB_; ++j) {
            float w = e[j] * inv;
            e[j] = w;
            const float* cvrow = cvf + ((size_t)h * NB_ + j) * DH_;
#pragma unroll
            for (int d = 0; d < DH_; ++d) acc[d] += w * cvrow[d];
        }
        float* op = out_c + ((size_t)h * N_ + i) * DH_;
#pragma unroll
        for (int d = 0; d < DH_; ++d) op[d] = acc[d];
        // importance (+2 for own block) and top-4 selection
        int own = i >> 5;
        e[own] += 2.0f;
        unsigned int used = 0;
        int* spo = sel + ((size_t)h * N_ + i) * S_;
        for (int t = 0; t < S_; ++t) {
            float best = -1e30f;
            int bj = 0;
            for (int j = 0; j < NB_; ++j) {
                bool ok = !(used & (1u << j));
                if (ok && e[j] > best) { best = e[j]; bj = j; }
            }
            used |= (1u << bj);
            spo[t] = bj;
        }
    }
}

// ---------------- fine branch: wave per (head, query), irregular block gather ----------------
__global__ void fine_kernel(const unsigned short* __restrict__ qh, const unsigned short* __restrict__ kh,
                            const unsigned short* __restrict__ vh, const int* __restrict__ sel,
                            float* __restrict__ out_f) {
    __shared__ float pSh[4][S_ * BC_];
    int wid  = (blockIdx.x * blockDim.x + threadIdx.x) >> 5;
    int lane = threadIdx.x & 31;
    int wib  = (threadIdx.x >> 5) & 3;
    if (wid >= H_ * N_) return;
    int h = wid >> 10, i = wid & (N_ - 1);
    const unsigned short* qp = qh + ((size_t)h * N_ + i) * DH_;
    float qreg[DH_];
#pragma unroll
    for (int d = 0; d < DH_; ++d) qreg[d] = bf2f(qp[d]);
    const int* sp = sel + ((size_t)h * N_ + i) * S_;
    int blks[S_];
#pragma unroll
    for (int t = 0; t < S_; ++t) blks[t] = sp[t];
    float p[S_];
    float mx = -1e30f;
#pragma unroll
    for (int t = 0; t < S_; ++t) {
        int j = blks[t] * BC_ + lane;
        const unsigned short* kp = kh + ((size_t)h * N_ + j) * DH_;
        float dot = 0.0f;
#pragma unroll
        for (int d = 0; d < DH_; ++d) dot += qreg[d] * bf2f(kp[d]);
        dot *= 0.125f;
        p[t] = (j <= i) ? dot : -1e30f;
        mx = fmaxf(mx, p[t]);
    }
    for (int off = 1; off < 32; off <<= 1) mx = fmaxf(mx, __shfl_xor(mx, off, 32));
    float sum = 0.0f;
#pragma unroll
    for (int t = 0; t < S_; ++t) {
        float e = (p[t] > -1e29f) ? __expf(p[t] - mx) : 0.0f;
        p[t] = e;
        sum += e;
    }
    for (int off = 1; off < 32; off <<= 1) sum += __shfl_xor(sum, off, 32);
    float inv = 1.0f / sum;
#pragma unroll
    for (int t = 0; t < S_; ++t) pSh[wib][t * BC_ + lane] = p[t] * inv;
    // role swap: lane accumulates output dims (lane, lane+32); same-wave DS in-order
    float a0 = 0.0f, a1 = 0.0f;
#pragma unroll
    for (int t = 0; t < S_; ++t) {
        int jb = blks[t] * BC_;
        for (int cc = 0; cc < BC_; ++cc) {
            float w = pSh[wib][t * BC_ + cc];
            const unsigned short* vp = vh + ((size_t)h * N_ + jb + cc) * DH_;
            a0 += w * bf2f(vp[lane]);
            a1 += w * bf2f(vp[lane + 32]);
        }
    }
    float* op = out_f + ((size_t)h * N_ + i) * DH_;
    op[lane] = a0;
    op[lane + 32] = a1;
}

// ---------------- sliding-window branch: wave-level flash attention with WMMA ----------------
__global__ void swin_kernel(const unsigned short* __restrict__ qh, const unsigned short* __restrict__ kh,
                            const unsigned short* __restrict__ vT, float* __restrict__ out_w) {
    __shared__ __align__(16) unsigned short ldsP[4][256];  // 16x16 bf16 P tile per wave
    int wid  = (blockIdx.x * blockDim.x + threadIdx.x) >> 5;
    int lane = threadIdx.x & 31;
    int wib  = (threadIdx.x >> 5) & 3;
    if (wid >= H_ * (N_ / 16)) return;
    int h = wid >> 6, qt = wid & 63, qlo = qt * 16;
    int lm = lane & 15, hi = lane >> 4;
    const unsigned short* kbase = kh + (size_t)h * N_ * DH_;
    const unsigned short* vbase = vT + (size_t)h * DH_ * N_;
    const unsigned short* qrow  = qh + ((size_t)h * N_ + qlo + lm) * DH_;
    bf16x16 aq0 = load_a_frag(qrow, hi);
    bf16x16 aq1 = load_a_frag(qrow + 32, hi);
    f32x8 accO[4] = {};
    float mrow[8], lrow[8];
#pragma unroll
    for (int r = 0; r < 8; ++r) { mrow[r] = -1e30f; lrow[r] = 0.0f; }
    int kmin = qlo - (W_ - 1);
    if (kmin < 0) kmin = 0;
    unsigned short* Pt = ldsP[wib];
    for (int jt = kmin >> 4; jt <= qt; ++jt) {
        const unsigned short* krow = kbase + (size_t)(jt * 16 + lm) * DH_;
        f32x8 s = {};
        s = wmma_bf16(aq0, load_b_frag(krow, hi), s);
        s = wmma_bf16(aq1, load_b_frag(krow + 32, hi), s);
        int j = jt * 16 + lm;   // this lane's key column in C-layout
        float sc[8], pr[8];
#pragma unroll
        for (int r = 0; r < 8; ++r) {
            int i = qlo + r + 8 * hi;
            int dd = i - j;
            sc[r] = (dd >= 0 && dd < W_) ? s[r] * 0.125f : -1e30f;
        }
#pragma unroll
        for (int r = 0; r < 8; ++r) {
            float v = sc[r];
            for (int off = 1; off < 16; off <<= 1) v = fmaxf(v, __shfl_xor(v, off, 32));
            float nm = fmaxf(mrow[r], v);
            float e = (sc[r] > -1e29f) ? __expf(sc[r] - nm) : 0.0f;
            float rs = e;
            for (int off = 1; off < 16; off <<= 1) rs += __shfl_xor(rs, off, 32);
            float corr = __expf(mrow[r] - nm);
            lrow[r] = lrow[r] * corr + rs;
            mrow[r] = nm;
            pr[r] = e;
#pragma unroll
            for (int t = 0; t < 4; ++t) accO[t][r] *= corr;
        }
        // C-layout P -> LDS (row-major 16x16 bf16); same-wave DS ops are in-order
#pragma unroll
        for (int r = 0; r < 8; ++r) Pt[(r + 8 * hi) * 16 + lm] = f2bf(pr[r]);
        // read back in A-fragment layout, zero-padding K=16..31
        u16x8 prow = *(const u16x8*)(Pt + lm * 16 + hi * 8);
        u16x16 pa = {};
#pragma unroll
        for (int e = 0; e < 8; ++e) pa[e] = prow[e];
        bf16x16 paf = __builtin_bit_cast(bf16x16, pa);
#pragma unroll
        for (int t = 0; t < 4; ++t) {
            u16x16 bu = {};   // B-fragment: elements 0..7 = k(hi*8 + 0..7); k=16..31 (elements 8..15) zero
            {
                const unsigned short* vrow = vbase + (size_t)(t * 16 + lm) * N_ + jt * 16 + hi * 8;
                u16x8 v0 = *(const u16x8*)(vrow);
#pragma unroll
                for (int e = 0; e < 8; ++e) { bu[e] = v0[e]; }
            }
            accO[t] = wmma_bf16(paf, __builtin_bit_cast(bf16x16, bu), accO[t]);
        }
    }
#pragma unroll
    for (int r = 0; r < 8; ++r) {
        int i = qlo + r + 8 * hi;
        float inv = 1.0f / lrow[r];
#pragma unroll
        for (int t = 0; t < 4; ++t)
            out_w[((size_t)h * N_ + i) * DH_ + t * 16 + lm] = accO[t][r] * inv;
    }
}

// ---------------- gates: sigmoid(h @ Wg) ----------------
__global__ void gates_kernel(const float* __restrict__ hf, const float* __restrict__ Wg,
                             float* __restrict__ gates) {
    int id = blockIdx.x * blockDim.x + threadIdx.x;
    if (id >= N_ * 24) return;
    int i = id / 24, c = id % 24;
    float acc = 0.0f;
    for (int k = 0; k < DIM_; ++k) acc += hf[(size_t)i * DIM_ + k] * Wg[(size_t)k * 24 + c];
    gates[id] = 1.0f / (1.0f + __expf(-acc));
}

// ---------------- gated combine -> bf16 for final GEMM ----------------
__global__ void combine_kernel(const float* __restrict__ gates, const float* __restrict__ oc,
                               const float* __restrict__ of_, const float* __restrict__ ow,
                               unsigned short* __restrict__ comb) {
    int id = blockIdx.x * blockDim.x + threadIdx.x;
    if (id >= N_ * HD_) return;
    int i = id >> 9, c = id & (HD_ - 1);
    int hh = c >> 6, d = c & 63;
    const float* g = gates + (size_t)i * 24 + hh * 3;
    size_t hoff = ((size_t)hh * N_ + i) * DH_ + d;
    comb[id] = f2bf(g[0] * oc[hoff] + g[1] * of_[hoff] + g[2] * ow[hoff]);
}

extern "C" void kernel_launch(void* const* d_in, const int* in_sizes, int n_in,
                              void* d_out, int out_size, void* d_ws, size_t ws_size,
                              hipStream_t stream) {
    (void)in_sizes; (void)n_in; (void)out_size; (void)ws_size;
    const float* x      = (const float*)d_in[0];
    const float* pos    = (const float*)d_in[1];
    const float* gamma  = (const float*)d_in[2];
    const float* Wp     = (const float*)d_in[3];
    const float* bp     = (const float*)d_in[4];
    const float* Wq     = (const float*)d_in[5];
    const float* Wk     = (const float*)d_in[6];
    const float* Wv     = (const float*)d_in[7];
    const float* Wo     = (const float*)d_in[8];
    const float* k_pos  = (const float*)d_in[9];
    const float* v_pos  = (const float*)d_in[10];
    const float* Wck    = (const float*)d_in[11];
    const float* bck    = (const float*)d_in[12];
    const float* Wcv    = (const float*)d_in[13];
    const float* bcv    = (const float*)d_in[14];
    const float* mem_ck = (const float*)d_in[15];
    const float* mem_cv = (const float*)d_in[16];
    const float* Wg     = (const float*)d_in[17];

    char* ws = (char*)d_ws;
    size_t off = 0;
    auto alloc = [&](size_t bytes) -> char* {
        char* p = ws + off;
        off = (off + bytes + 255) & ~(size_t)255;
        return p;
    };
    float*          hf     = (float*)alloc((size_t)N_ * DIM_ * 4);
    unsigned short* hbf    = (unsigned short*)alloc((size_t)N_ * DIM_ * 2);
    unsigned short* WqT    = (unsigned short*)alloc((size_t)DIM_ * HD_ * 2);
    unsigned short* WkT    = (unsigned short*)alloc((size_t)DIM_ * HD_ * 2);
    unsigned short* WvT    = (unsigned short*)alloc((size_t)DIM_ * HD_ * 2);
    unsigned short* WoT    = (unsigned short*)alloc((size_t)HD_ * DIM_ * 2);
    unsigned short* WckT   = (unsigned short*)alloc((size_t)BC_ * DH_ * DH_ * 2);
    unsigned short* WcvT   = (unsigned short*)alloc((size_t)BC_ * DH_ * DH_ * 2);
    float*          qf     = (float*)alloc((size_t)N_ * HD_ * 4);
    float*          kf     = (float*)alloc((size_t)N_ * HD_ * 4);
    float*          vf     = (float*)alloc((size_t)N_ * HD_ * 4);
    unsigned short* qhB    = (unsigned short*)alloc((size_t)H_ * N_ * DH_ * 2);
    unsigned short* khB    = (unsigned short*)alloc((size_t)H_ * N_ * DH_ * 2);
    unsigned short* vhB    = (unsigned short*)alloc((size_t)H_ * N_ * DH_ * 2);
    unsigned short* vTB    = (unsigned short*)alloc((size_t)H_ * DH_ * N_ * 2);
    unsigned short* kbflat = (unsigned short*)alloc((size_t)H_ * NB_ * BC_ * DH_ * 2);
    unsigned short* vbflat = (unsigned short*)alloc((size_t)H_ * NB_ * BC_ * DH_ * 2);
    float*          ckf    = (float*)alloc((size_t)H_ * NB_ * DH_ * 4);
    float*          cvfb   = (float*)alloc((size_t)H_ * NB_ * DH_ * 4);
    unsigned short* ckbf16 = (unsigned short*)alloc((size_t)H_ * NB_ * DH_ * 2);
    float*          out_c  = (float*)alloc((size_t)H_ * N_ * DH_ * 4);
    float*          out_f  = (float*)alloc((size_t)H_ * N_ * DH_ * 4);
    float*          out_w  = (float*)alloc((size_t)H_ * N_ * DH_ * 4);
    int*            selbuf = (int*)alloc((size_t)H_ * N_ * S_ * 4);
    float*          gatesb = (float*)alloc((size_t)N_ * 24 * 4);
    unsigned short* comb   = (unsigned short*)alloc((size_t)N_ * HD_ * 2);

    // 1) prep: positional add + RMSNorm
    prep_kernel<<<N_, 256, 0, stream>>>(x, pos, gamma, Wp, bp, hf, hbf);

    // 2) weight transposes to bf16 B^T layouts
    transpose_bf16_kernel<<<(DIM_ * HD_ + 255) / 256, 256, 0, stream>>>(Wq, WqT, DIM_, HD_);
    transpose_bf16_kernel<<<(DIM_ * HD_ + 255) / 256, 256, 0, stream>>>(Wk, WkT, DIM_, HD_);
    transpose_bf16_kernel<<<(DIM_ * HD_ + 255) / 256, 256, 0, stream>>>(Wv, WvT, DIM_, HD_);
    transpose_bf16_kernel<<<(HD_ * DIM_ + 255) / 256, 256, 0, stream>>>(Wo, WoT, HD_, DIM_);
    transpose_bf16_kernel<<<(BC_ * DH_ * DH_ + 255) / 256, 256, 0, stream>>>(Wck, WckT, BC_ * DH_, DH_);
    transpose_bf16_kernel<<<(BC_ * DH_ * DH_ + 255) / 256, 256, 0, stream>>>(Wcv, WcvT, BC_ * DH_, DH_);

    // 3) QKV projections (WMMA): 2048 tiles each -> 512 blocks x 4 waves
    gemm_bf16_wmma_kernel<<<512, 128, 0, stream>>>(hbf, WqT, nullptr, qf, N_, HD_, DIM_);
    gemm_bf16_wmma_kernel<<<512, 128, 0, stream>>>(hbf, WkT, nullptr, kf, N_, HD_, DIM_);
    gemm_bf16_wmma_kernel<<<512, 128, 0, stream>>>(hbf, WvT, nullptr, vf, N_, HD_, DIM_);

    // 4) per-head repack + (k+k_pos)/(v+v_pos) block-flattening
    qkv_repack_kernel<<<(N_ * HD_ + 255) / 256, 256, 0, stream>>>(qf, kf, vf, k_pos, v_pos,
                                                                  qhB, khB, vhB, vTB, kbflat, vbflat);

    // 5) block compression GEMMs (WMMA): M=256, N=64, K=2048 -> 64 tiles
    gemm_bf16_wmma_kernel<<<16, 128, 0, stream>>>(kbflat, WckT, bck, ckf, H_ * NB_, DH_, BC_ * DH_);
    gemm_bf16_wmma_kernel<<<16, 128, 0, stream>>>(vbflat, WcvT, bcv, cvfb, H_ * NB_, DH_, BC_ * DH_);
    cvt_bf16_kernel<<<(H_ * NB_ * DH_ + 255) / 256, 256, 0, stream>>>(ckf, ckbf16, H_ * NB_ * DH_);

    // 6) compressed attention + top-4 block selection (WMMA scores)
    compressed_kernel<<<128, 128, 0, stream>>>(qhB, ckbf16, cvfb, mem_ck, mem_cv, out_c, selbuf);

    // 7) fine branch (irregular gather, wave per query)
    fine_kernel<<<2048, 128, 0, stream>>>(qhB, khB, vhB, selbuf, out_f);

    // 8) sliding-window flash attention (WMMA QK^T and PV)
    swin_kernel<<<128, 128, 0, stream>>>(qhB, khB, vTB, out_w);

    // 9) gates + gated combine
    gates_kernel<<<(N_ * 24 + 255) / 256, 256, 0, stream>>>(hf, Wg, gatesb);
    combine_kernel<<<(N_ * HD_ + 255) / 256, 256, 0, stream>>>(gatesb, out_c, out_f, out_w, comb);

    // 10) final output projection (WMMA) straight into d_out
    gemm_bf16_wmma_kernel<<<512, 128, 0, stream>>>(comb, WoT, nullptr, (float*)d_out, N_, DIM_, HD_);
}
